// ChargeModel_41180146434459
// MI455X (gfx1250) — compile-verified, matrix operand order checked by reading, not run on hardware
//
#include <hip/hip_runtime.h>
#include <hip/hip_bf16.h>

// 3-layer GCN over 4000 independent 64-node graphs.
// Per graph:  H <- Adense(64x64) @ (H @ W) + b  (x3), then mean -> scalar.
// All matmuls via v_wmma_f32_16x16x32_f16 (f16 in, f32 accumulate).

typedef __attribute__((ext_vector_type(16))) _Float16 v16h;
typedef __attribute__((ext_vector_type(8)))  float    v8f;

#define STR  72          // padded LDS row stride in halves (144B, 16B aligned)
#define EPG  512         // edges per graph
#define NTHR 256         // 8 waves of 32
#define WOFF 34816       // byte offset of first transposed weight matrix
#define WSZ  9216        // bytes per staged f16 matrix (64 rows x 72 halves)

// Load a 16x32 f16 fragment (A-operand layout, ISA 7.12.2) from row-major LDS.
// Same loader serves the B operand because B matrices are staged transposed
// (rows = N, cols = K), matching the ISA's column-major B expectation.
__device__ __forceinline__ v16h ld_frag(const _Float16* __restrict__ base,
                                        int row0, int k0, int lane) {
  const int h = lane >> 4;   // 0: K-offsets {0..7,16..23}; 1: {8..15,24..31}
  const int r = lane & 15;   // row within tile
  const _Float16* p = base + (row0 + r) * STR + k0 + h * 8;
  v16h f;
#pragma unroll
  for (int i = 0; i < 8; ++i) f[i] = p[i];          // VGPR 0..3
#pragma unroll
  for (int i = 0; i < 8; ++i) f[8 + i] = p[16 + i]; // VGPR 4..7
  return f;
}

// One 16x16 output tile of a 64x64x64 matmul: 2 x wmma over K=64.
__device__ __forceinline__ v8f mm_tile(const _Float16* __restrict__ A,
                                       const _Float16* __restrict__ B,
                                       int mi, int ni, int lane) {
  v8f c = {};
  v16h a0 = ld_frag(A, mi, 0,  lane);
  v16h b0 = ld_frag(B, ni, 0,  lane);
  c = __builtin_amdgcn_wmma_f32_16x16x32_f16(false, a0, false, b0,
                                             (short)0, c, false, false);
  v16h a1 = ld_frag(A, mi, 32, lane);
  v16h b1 = ld_frag(B, ni, 32, lane);
  c = __builtin_amdgcn_wmma_f32_16x16x32_f16(false, a1, false, b1,
                                             (short)0, c, false, false);
  return c;
}

__global__ __launch_bounds__(NTHR)
void gcn3_wmma_kernel(const float* __restrict__ x,
                      const int*   __restrict__ esrc,
                      const int*   __restrict__ edst,
                      const float* __restrict__ W1, const float* __restrict__ b1,
                      const float* __restrict__ W2, const float* __restrict__ b2,
                      const float* __restrict__ W3, const float* __restrict__ b3,
                      float* __restrict__ y) {
  // ---- hand-packed LDS (64,768 B total) ----
  __shared__ __align__(16) unsigned char smem[64768];
  float*    A32 = (float*)(smem);                 // 16384 B, setup only
  _Float16* Tt  = (_Float16*)(smem);              //  9216 B, layers (reuses A32)
  _Float16* Ah  = (_Float16*)(smem + 16384);      //  9216 B, dense-adj f16
  _Float16* Hh  = (_Float16*)(smem + 25600);      //  9216 B, activations f16
  _Float16* Wt0 = (_Float16*)(smem + WOFF);           // transposed W1
  _Float16* Wt1 = (_Float16*)(smem + WOFF + WSZ);     // transposed W2
  _Float16* Wt2 = (_Float16*)(smem + WOFF + 2 * WSZ); // transposed W3
  int*   deg  = (int*)  (smem + 62464);           //   256 B
  float* dinv = (float*)(smem + 62720);           //   256 B
  float* bsh  = (float*)(smem + 62976);           //   768 B (b1|b2|b3)
  float* red  = (float*)(smem + 63744);           //  1024 B

  const int g    = blockIdx.x;
  const int tid  = threadIdx.x;
  const int lane = tid & 31;
  const int wv   = tid >> 5;

  // per-wave tiles: wave wv owns tiles wv and wv+8 of the 4x4 tile grid
  const int mi0 = (wv >> 2) * 16,       ni0 = (wv & 3) * 16;
  const int mi1 = ((wv + 8) >> 2) * 16, ni1 = ((wv + 8) & 3) * 16;
  // C/D fragment lane mapping (ISA 7.12.2): element (M,N) = (8*hh + v, n)
  const int n_cd  = lane & 15;
  const int hh_cd = lane >> 4;

  // ---- setup: zero adj, deg=1 (self-loop), stage biases ----
  for (int i = tid; i < 4096; i += NTHR) A32[i] = 0.0f;
  if (tid < 64) {
    deg[tid]      = 1;
    bsh[tid]      = b1[tid];
    bsh[64 + tid] = b2[tid];
    bsh[128+ tid] = b3[tid];
  }
  // stage weights transposed (rows = out-channel N, cols = in-channel K)
  for (int i = tid; i < 4096; i += NTHR) {
    const int k = i >> 6, n = i & 63;
    Wt0[n * STR + k] = (_Float16)W1[i];
    Wt1[n * STR + k] = (_Float16)W2[i];
    Wt2[n * STR + k] = (_Float16)W3[i];
  }
  // stage this graph's node features (row-major, nodes x feats)
  const float* xg = x + (size_t)g * 64 * 64;
  for (int i = tid; i < 4096; i += NTHR)
    Hh[(i >> 6) * STR + (i & 63)] = (_Float16)xg[i];
  __syncthreads();

  // ---- degrees over dst (+ implicit self-loop via init=1) ----
  const int eb = g * EPG;
  const int s0 = esrc[eb + tid]        & 63;   // edges are block-local
  const int d0 = edst[eb + tid]        & 63;
  const int s1 = esrc[eb + NTHR + tid] & 63;
  const int d1 = edst[eb + NTHR + tid] & 63;
  atomicAdd(&deg[d0], 1);
  atomicAdd(&deg[d1], 1);
  __syncthreads();
  if (tid < 64) dinv[tid] = rsqrtf((float)deg[tid]);   // deg >= 1 always
  __syncthreads();

  // ---- dense normalized adjacency: A[dst][src] += dinv[src]*dinv[dst] ----
  atomicAdd(&A32[d0 * 64 + s0], dinv[s0] * dinv[d0]);
  atomicAdd(&A32[d1 * 64 + s1], dinv[s1] * dinv[d1]);
  if (tid < 64) atomicAdd(&A32[tid * 64 + tid], dinv[tid] * dinv[tid]);
  __syncthreads();
  for (int i = tid; i < 4096; i += NTHR)
    Ah[(i >> 6) * STR + (i & 63)] = (_Float16)A32[i];
  __syncthreads();   // after this, A32 region is reused as Tt

  // ---- 3 GCN layers; 16 output tiles, 2 per wave (straight-line) ----
  float acc_sum = 0.0f;
#pragma unroll 1
  for (int layer = 0; layer < 3; ++layer) {
    const _Float16* W = (const _Float16*)(smem + WOFF + layer * WSZ);

    // matmul 1: T = H @ W, stored transposed into Tt (rows = channel).
    // Two independent accumulator chains per wave -> WMMA hazard overlap.
    {
      v8f c0 = mm_tile(Hh, W, mi0, ni0, lane);
      v8f c1 = mm_tile(Hh, W, mi1, ni1, lane);
#pragma unroll
      for (int v = 0; v < 8; ++v) {
        const int m = hh_cd * 8 + v;
        Tt[(ni0 + n_cd) * STR + (mi0 + m)] = (_Float16)c0[v];
        Tt[(ni1 + n_cd) * STR + (mi1 + m)] = (_Float16)c1[v];
      }
    }
    __syncthreads();

    // matmul 2: H' = Ah @ T (+ bias); layer 3 feeds the readout directly
    const float* bb = bsh + layer * 64;
    {
      v8f c0 = mm_tile(Ah, Tt, mi0, ni0, lane);
      v8f c1 = mm_tile(Ah, Tt, mi1, ni1, lane);
      const float bv0 = bb[ni0 + n_cd];
      const float bv1 = bb[ni1 + n_cd];
      if (layer < 2) {
#pragma unroll
        for (int v = 0; v < 8; ++v) {
          const int m = hh_cd * 8 + v;
          Hh[(mi0 + m) * STR + (ni0 + n_cd)] = (_Float16)(c0[v] + bv0);
          Hh[(mi1 + m) * STR + (ni1 + n_cd)] = (_Float16)(c1[v] + bv1);
        }
      } else {
#pragma unroll
        for (int v = 0; v < 8; ++v) acc_sum += (c0[v] + bv0) + (c1[v] + bv1);
      }
    }
    __syncthreads();
  }

  // ---- per-graph mean over 64x64 elements ----
  red[tid] = acc_sum;
  __syncthreads();
#pragma unroll
  for (int s = 128; s > 0; s >>= 1) {
    if (tid < s) red[tid] += red[tid + s];
    __syncthreads();
  }
  if (tid == 0) y[g] = red[0] * (1.0f / 4096.0f);
}

extern "C" void kernel_launch(void* const* d_in, const int* in_sizes, int n_in,
                              void* d_out, int out_size, void* d_ws, size_t ws_size,
                              hipStream_t stream) {
  (void)n_in; (void)d_ws; (void)ws_size;
  const float* x  = (const float*)d_in[0];
  const int*   ei = (const int*)  d_in[1];     // [2, E] stacked: src then dst
  const float* W1 = (const float*)d_in[2];
  const float* b1 = (const float*)d_in[3];
  const float* W2 = (const float*)d_in[4];
  const float* b2 = (const float*)d_in[5];
  const float* W3 = (const float*)d_in[6];
  const float* b3 = (const float*)d_in[7];
  float* y = (float*)d_out;

  const int ne  = in_sizes[1] / 2;             // total edges (B * 512)
  const int*   src = ei;
  const int*   dst = ei + ne;
  const int B = out_size;                      // 4000 graphs

  gcn3_wmma_kernel<<<B, NTHR, 0, stream>>>(x, src, dst,
                                           W1, b1, W2, b2, W3, b3, y);
}
